// RWKV_57604101374168
// MI455X (gfx1250) — compile-verified, matrix operand order checked by reading, not run on hardware
//
#include <hip/hip_runtime.h>
#include <cmath>

// ---------------------------------------------------------------------------
// RWKV6 forward for MI455X (gfx1250, wave32, WMMA bf16, async LDS staging)
// ---------------------------------------------------------------------------

typedef __attribute__((ext_vector_type(16))) __bf16 v16bf;
typedef __attribute__((ext_vector_type(8)))  __bf16 v8bf;
typedef __attribute__((ext_vector_type(4)))  __bf16 v4bf;
typedef __attribute__((ext_vector_type(8)))  float  v8f;
typedef __attribute__((ext_vector_type(4)))  float  v4f;

constexpr int Bn   = 4;
constexpr int Tn   = 512;
constexpr int Mn   = Bn * Tn;     // 2048 rows
constexpr int Cn   = 1024;
constexpr int Hn   = 16;
constexpr int Kn   = 64;          // head dim
constexpr int DHn  = 3584;
constexpr int NV   = 50257;

// ---------------------------------------------------------------------------
// fp32 -> bf16 conversion, 4 elements/thread (n must be a multiple of 4)
// ---------------------------------------------------------------------------
__global__ void cvt_bf16_kernel(const float* __restrict__ in, __bf16* __restrict__ out, size_t n) {
    size_t i4 = ((size_t)blockIdx.x * 256 + threadIdx.x) * 4;
    if (i4 >= n) return;
    v4f x = *(const v4f*)(in + i4);
    *(v4bf*)(out + i4) = __builtin_convertvector(x, v4bf);
}

// ---------------------------------------------------------------------------
// fp32 (Kt x Nt) -> bf16 (Kt x Np) with zero padding in [Nt, Np)
// Np multiple of 128 so GEMM loads are 16B-aligned & branch-free.
// ---------------------------------------------------------------------------
__global__ void cvt_pad_bf16_kernel(const float* __restrict__ in, __bf16* __restrict__ out,
                                    int Kt, int Nt, int Np)
{
    size_t i4 = ((size_t)blockIdx.x * 256 + threadIdx.x) * 4;
    if (i4 >= (size_t)Kt * Np) return;
    int r = (int)(i4 / Np);
    int c = (int)(i4 % Np);
    v4bf o;
    #pragma unroll
    for (int e = 0; e < 4; ++e) {
        int cc = c + e;
        o[e] = (cc < Nt) ? (__bf16)in[(size_t)r * Nt + cc] : (__bf16)0.0f;
    }
    *(v4bf*)(out + i4) = o;
}

// ---------------------------------------------------------------------------
// bf16 WMMA GEMM: Y(Mt x Nt) = A(Mt x Kt) * W(Kt x Np, padded), fp32 acc.
// Block: 256 threads (8 waves), tile 64(M) x 128(N), K-step 32.
// Wave (wm, wn) in 2x4; each wave: 2x2 grid of 16x16 WMMA tiles.
// A staged row-major via global_load_async_to_lds_b128 (ASYNCcnt path).
// B staged transposed (Bst[n][k]) so fragments are 2x ds_load_b128 each.
// ---------------------------------------------------------------------------
__global__ __launch_bounds__(256) void gemm_bf16_wmma(
    const __bf16* __restrict__ A, const __bf16* __restrict__ W,
    float* __restrict__ Y, int Mt, int Kt, int Nt, int Np)
{
    __shared__ __bf16 As[64 * 32];    // 4 KB
    __shared__ __bf16 Bst[128 * 32];  // 8 KB, transposed: Bst[n][k]
    const int tid  = threadIdx.x;
    const int lane = tid & 31;
    const int wave = tid >> 5;
    const int wm   = wave >> 2;       // 0..1
    const int wn   = wave & 3;        // 0..3
    const int bm   = blockIdx.y * 64;
    const int bn   = blockIdx.x * 128;
    const int half = lane >> 4;
    const int q    = lane & 15;

    v8f acc[2][2] = {};

    for (int kk = 0; kk < Kt; kk += 32) {
        // ---- stage A tile (64 x 32): one async b128 chunk per thread ----
        {
            int r  = tid >> 2;        // 0..63
            int cc = tid & 3;         // 0..3 (chunks of 8 bf16)
            const __bf16* gp = A + (size_t)(bm + r) * Kt + kk + cc * 8;
            unsigned lds_off = (unsigned)(size_t)(&As[r * 32 + cc * 8]);
            asm volatile("global_load_async_to_lds_b128 %0, %1, off"
                         :: "v"(lds_off), "v"((unsigned long long)(size_t)gp)
                         : "memory");
        }
        // ---- stage B tile (32 x 128) transposed: b128 load + b16 scatter ----
        #pragma unroll
        for (int ch = 0; ch < 2; ++ch) {
            int j  = tid + ch * 256;  // 0..511
            int r  = j >> 4;          // k row 0..31
            int cc = j & 15;          // n chunk 0..15
            v8bf vb = *(const v8bf*)(W + (size_t)(kk + r) * Np + bn + cc * 8);
            #pragma unroll
            for (int e = 0; e < 8; ++e)
                Bst[(cc * 8 + e) * 32 + r] = vb[e];
        }
        asm volatile("s_wait_asynccnt 0x0" ::: "memory");
        __syncthreads();

        // ---- build fragments (two contiguous b128 chunks each) ----
        v16bf afrag[2], bfrag[2];
        #pragma unroll
        for (int mt = 0; mt < 2; ++mt) {
            const __bf16* ap = &As[(wm * 32 + mt * 16 + q) * 32];
            v8bf lo = *(const v8bf*)(ap + half * 8);
            v8bf hi = *(const v8bf*)(ap + 16 + half * 8);
            afrag[mt] = __builtin_shufflevector(lo, hi, 0, 1, 2, 3, 4, 5, 6, 7,
                                                        8, 9, 10, 11, 12, 13, 14, 15);
        }
        #pragma unroll
        for (int nt = 0; nt < 2; ++nt) {
            const __bf16* bp = &Bst[(wn * 32 + nt * 16 + q) * 32];
            v8bf lo = *(const v8bf*)(bp + half * 8);
            v8bf hi = *(const v8bf*)(bp + 16 + half * 8);
            bfrag[nt] = __builtin_shufflevector(lo, hi, 0, 1, 2, 3, 4, 5, 6, 7,
                                                        8, 9, 10, 11, 12, 13, 14, 15);
        }
        #pragma unroll
        for (int mt = 0; mt < 2; ++mt)
            #pragma unroll
            for (int nt = 0; nt < 2; ++nt)
                acc[mt][nt] = __builtin_amdgcn_wmma_f32_16x16x32_bf16(
                    false, afrag[mt], false, bfrag[nt], (short)0, acc[mt][nt], false, false);
        __syncthreads();
    }

    // ---- store: VGPR v -> row v (lanes 0-15) / v+8 (lanes 16-31), col = q ----
    #pragma unroll
    for (int mt = 0; mt < 2; ++mt) {
        #pragma unroll
        for (int nt = 0; nt < 2; ++nt) {
            int col = bn + wn * 32 + nt * 16 + q;
            if (col < Nt) {
                #pragma unroll
                for (int v = 0; v < 8; ++v) {
                    int row = bm + wm * 32 + mt * 16 + ((lane < 16) ? v : v + 8);
                    Y[(size_t)row * Nt + col] = acc[mt][nt][v];
                }
            }
        }
    }
}

// ---------------------------------------------------------------------------
// Embedding gather + LayerNorm (one block per token row)
// ---------------------------------------------------------------------------
__global__ __launch_bounds__(256) void embed_ln_kernel(
    const int* __restrict__ tokens, const float* __restrict__ embed,
    const float* __restrict__ w, const float* __restrict__ b, float* __restrict__ y)
{
    __shared__ float red[256];
    int m = blockIdx.x, tid = threadIdx.x;
    const float* row = embed + (size_t)tokens[m] * Cn;
    float s = 0.f;
    for (int c = tid; c < Cn; c += 256) s += row[c];
    red[tid] = s; __syncthreads();
    for (int st = 128; st > 0; st >>= 1) { if (tid < st) red[tid] += red[tid + st]; __syncthreads(); }
    float mu = red[0] / Cn; __syncthreads();
    float v2 = 0.f;
    for (int c = tid; c < Cn; c += 256) { float d = row[c] - mu; v2 += d * d; }
    red[tid] = v2; __syncthreads();
    for (int st = 128; st > 0; st >>= 1) { if (tid < st) red[tid] += red[tid + st]; __syncthreads(); }
    float rs = rsqrtf(red[0] / Cn + 1e-5f);
    for (int c = tid; c < Cn; c += 256)
        y[(size_t)m * Cn + c] = (row[c] - mu) * rs * w[c] + b[c];
}

// ---------------------------------------------------------------------------
// LayerNorm (one block per row)
// ---------------------------------------------------------------------------
__global__ __launch_bounds__(256) void ln_kernel(
    const float* __restrict__ x, const float* __restrict__ w,
    const float* __restrict__ b, float* __restrict__ y)
{
    __shared__ float red[256];
    int m = blockIdx.x, tid = threadIdx.x;
    const float* row = x + (size_t)m * Cn;
    float s = 0.f;
    for (int c = tid; c < Cn; c += 256) s += row[c];
    red[tid] = s; __syncthreads();
    for (int st = 128; st > 0; st >>= 1) { if (tid < st) red[tid] += red[tid + st]; __syncthreads(); }
    float mu = red[0] / Cn; __syncthreads();
    float v2 = 0.f;
    for (int c = tid; c < Cn; c += 256) { float d = row[c] - mu; v2 += d * d; }
    red[tid] = v2; __syncthreads();
    for (int st = 128; st > 0; st >>= 1) { if (tid < st) red[tid] += red[tid + st]; __syncthreads(); }
    float rs = rsqrtf(red[0] / Cn + 1e-5f);
    for (int c = tid; c < Cn; c += 256)
        y[(size_t)m * Cn + c] = (row[c] - mu) * rs * w[c] + b[c];
}

// ---------------------------------------------------------------------------
// x_pre = xn + (x_shift - xn) * premix   (x_shift = previous t, zero at t==0)
// ---------------------------------------------------------------------------
__global__ void premix_kernel(const float* __restrict__ xn, const float* __restrict__ premix,
                              float* __restrict__ xpre)
{
    size_t idx = (size_t)blockIdx.x * 256 + threadIdx.x;
    if (idx >= (size_t)Mn * Cn) return;
    int c = (int)(idx % Cn);
    size_t m = idx / Cn;
    int t = (int)(m % Tn);
    float xv = xn[idx];
    float sh = (t == 0) ? 0.f : xn[idx - Cn];
    xpre[idx] = xv + (sh - xv) * premix[c];
}

// ---------------------------------------------------------------------------
// LoRA stage A: T[m, j] = tanh( X[m,:] @ Wa[:, j] ), j < D (D = 32 or 64)
// ---------------------------------------------------------------------------
__global__ __launch_bounds__(64) void lora_a_tanh_kernel(
    const float* __restrict__ X, const float* __restrict__ Wa,
    float* __restrict__ Tout, int D)
{
    __shared__ float xrow[Cn];
    int m = blockIdx.x;
    for (int c4 = threadIdx.x * 4; c4 < Cn; c4 += 64 * 4)
        *(v4f*)(xrow + c4) = *(const v4f*)(X + (size_t)m * Cn + c4);
    __syncthreads();
    if (threadIdx.x < D) {
        float acc = 0.f;
        for (int c = 0; c < Cn; ++c) acc += xrow[c] * Wa[(size_t)c * D + threadIdx.x];
        Tout[(size_t)m * 64 + threadIdx.x] = tanhf(acc);
    }
}

// ---------------------------------------------------------------------------
// DDLerp stage B: dd = xn + (shift - xn) * (base + T @ Wb)
// ---------------------------------------------------------------------------
__global__ void ddlerp_b_kernel(const float* __restrict__ xn, const float* __restrict__ Tm,
                                const float* __restrict__ base, const float* __restrict__ Wb,
                                float* __restrict__ dd, int D)
{
    size_t idx = (size_t)blockIdx.x * 256 + threadIdx.x;
    if (idx >= (size_t)Mn * Cn) return;
    int c = (int)(idx % Cn);
    size_t m = idx / Cn;
    int t = (int)(m % Tn);
    float y = base[c];
    for (int j = 0; j < D; ++j) y += Tm[m * 64 + j] * Wb[(size_t)j * Cn + c];
    float xv = xn[idx];
    float sh = (t == 0) ? 0.f : xn[idx - Cn];
    dd[idx] = xv + (sh - xv) * y;
}

// ---------------------------------------------------------------------------
// decay: w = exp(-exp(base + T @ Wb)), D = 64
// ---------------------------------------------------------------------------
__global__ void decay_kernel(const float* __restrict__ Tm, const float* __restrict__ base,
                             const float* __restrict__ Wb, float* __restrict__ wout)
{
    size_t idx = (size_t)blockIdx.x * 256 + threadIdx.x;
    if (idx >= (size_t)Mn * Cn) return;
    int c = (int)(idx % Cn);
    size_t m = idx / Cn;
    float y = base[c];
    #pragma unroll 8
    for (int j = 0; j < 64; ++j) y += Tm[m * 64 + j] * Wb[(size_t)j * Cn + c];
    wout[idx] = expf(-expf(y));
}

// ---------------------------------------------------------------------------
// WKV linear-attention scan. One block per (batch, head); 64 threads.
// Thread n owns state column kv[:, n] in 64 registers.
// ---------------------------------------------------------------------------
__global__ __launch_bounds__(64) void wkv_kernel(
    const float* __restrict__ r, const float* __restrict__ k,
    const float* __restrict__ v, const float* __restrict__ w,
    const float* __restrict__ u, float* __restrict__ out)
{
    int bh = blockIdx.x;
    int b = bh / Hn, h = bh % Hn;
    int n = threadIdx.x;
    __shared__ float rs[Kn], ks[Kn], wsah[Kn], us[Kn];
    us[n] = u[h * Kn + n];
    float s[Kn];
    #pragma unroll
    for (int i = 0; i < Kn; ++i) s[i] = 0.f;

    for (int t = 0; t < Tn; ++t) {
        size_t off = ((size_t)(b * Tn + t)) * Cn + h * Kn;
        __syncthreads();
        rs[n]   = r[off + n];
        ks[n]   = k[off + n];
        wsah[n] = w[off + n];
        __syncthreads();
        float vn = v[off + n];
        float o = 0.f;
        #pragma unroll
        for (int i = 0; i < Kn; ++i) {
            float kvu = ks[i] * vn;
            o += rs[i] * (s[i] + kvu * us[i]);
            s[i] = s[i] * wsah[i] + kvu;
        }
        out[off + n] = o;
    }
}

// ---------------------------------------------------------------------------
// GroupNorm (in place): grid (Mn, Hn), 64 threads, eps = 0.00064
// ---------------------------------------------------------------------------
__global__ __launch_bounds__(64) void groupnorm_kernel(
    float* __restrict__ x, const float* __restrict__ w, const float* __restrict__ b)
{
    int m = blockIdx.x, g = blockIdx.y, n = threadIdx.x;
    size_t off = (size_t)m * Cn + g * Kn + n;
    float v = x[off];
    __shared__ float red[Kn];
    red[n] = v; __syncthreads();
    for (int st = 32; st > 0; st >>= 1) { if (n < st) red[n] += red[n + st]; __syncthreads(); }
    float mu = red[0] / Kn; __syncthreads();
    float d = v - mu;
    red[n] = d * d; __syncthreads();
    for (int st = 32; st > 0; st >>= 1) { if (n < st) red[n] += red[n + st]; __syncthreads(); }
    float rs = rsqrtf(red[0] / Kn + 0.00064f);
    int c = g * Kn + n;
    x[off] = d * rs * w[c] + b[c];
}

// ---------------------------------------------------------------------------
// Misc elementwise
// ---------------------------------------------------------------------------
__global__ void silu_mul_kernel(float* __restrict__ out, const float* __restrict__ gate, size_t n) {
    size_t i = (size_t)blockIdx.x * 256 + threadIdx.x;
    if (i >= n) return;
    float g = gate[i];
    out[i] *= g / (1.f + expf(-g));
}

__global__ void add_inplace_kernel(float* __restrict__ x, const float* __restrict__ y, size_t n) {
    size_t i = (size_t)blockIdx.x * 256 + threadIdx.x;
    if (i < n) x[i] += y[i];
}

__global__ void cm_lerp_kernel(const float* __restrict__ xn, const float* __restrict__ lin,
                               const float* __restrict__ lgate, float* __restrict__ inp,
                               float* __restrict__ gate2)
{
    size_t idx = (size_t)blockIdx.x * 256 + threadIdx.x;
    if (idx >= (size_t)Mn * Cn) return;
    int c = (int)(idx % Cn);
    size_t m = idx / Cn;
    int t = (int)(m % Tn);
    float xv = xn[idx];
    float sh = (t == 0) ? 0.f : xn[idx - Cn];
    inp[idx]   = xv + (sh - xv) * lin[c];
    gate2[idx] = xv + (sh - xv) * lgate[c];
}

__global__ void relusq_kernel(float* __restrict__ h, size_t n) {
    size_t i = (size_t)blockIdx.x * 256 + threadIdx.x;
    if (i >= n) return;
    float v = fmaxf(h[i], 0.f);
    h[i] = v * v;
}

__global__ void cm_final_kernel(float* __restrict__ x, const float* __restrict__ hout,
                                const float* __restrict__ gg, size_t n)
{
    size_t i = (size_t)blockIdx.x * 256 + threadIdx.x;
    if (i >= n) return;
    x[i] += hout[i] / (1.f + expf(-gg[i]));
}

// ---------------------------------------------------------------------------
// Host-side GEMM wrapper: bf16 staging (padded W) + WMMA GEMM
// ---------------------------------------------------------------------------
static void run_gemm(hipStream_t stream, const float* A, const float* W, float* Y,
                     int Mt, int Kt, int Nt, __bf16* abf, __bf16* wbf)
{
    int Np = (Nt + 127) & ~127;
    size_t na = (size_t)Mt * Kt;
    size_t nw = (size_t)Kt * Np;
    cvt_bf16_kernel<<<(unsigned)((na / 4 + 255) / 256), 256, 0, stream>>>(A, abf, na);
    cvt_pad_bf16_kernel<<<(unsigned)((nw / 4 + 255) / 256), 256, 0, stream>>>(W, wbf, Kt, Nt, Np);
    dim3 grid(Np / 128, Mt / 64);
    gemm_bf16_wmma<<<grid, 256, 0, stream>>>(abf, wbf, Y, Mt, Kt, Nt, Np);
}

// ---------------------------------------------------------------------------
// Workspace layout (bytes). Total ~226 MB.
// ---------------------------------------------------------------------------
constexpr size_t SZ_MC = (size_t)Mn * Cn * sizeof(float);         // 8 MB
constexpr size_t o_x   = 0;
constexpr size_t o_xn  = o_x  + SZ_MC;
constexpr size_t o_dd  = o_xn + SZ_MC;                            // 5 x 8 MB
constexpr size_t o_p   = o_dd + 5 * SZ_MC;                        // 5 x 8 MB (also h: M*DH*4 = 28 MB)
constexpr size_t o_lt  = o_p  + 5 * SZ_MC;                        // M*64*4
constexpr size_t o_tmp = o_lt + (size_t)Mn * 64 * sizeof(float);  // xpre / Wo output
constexpr size_t o_out = o_tmp + SZ_MC;                           // wkv out
constexpr size_t o_abf = o_out + SZ_MC;                           // M*DH bf16
constexpr size_t o_wbf = o_abf + (size_t)Mn * DHn * 2;            // C * pad128(NV) bf16

extern "C" void kernel_launch(void* const* d_in, const int* in_sizes, int n_in,
                              void* d_out, int out_size, void* d_ws, size_t ws_size,
                              hipStream_t stream)
{
    (void)in_sizes; (void)n_in; (void)out_size; (void)ws_size;

    const int*   tokens  = (const int*)d_in[0];
    const float* embed   = (const float*)d_in[1];
    const float* elnw    = (const float*)d_in[2];
    const float* elnb    = (const float*)d_in[3];
    const float* olnw    = (const float*)d_in[4];
    const float* olnb    = (const float*)d_in[5];
    const float* unembed = (const float*)d_in[6];

    char* ws = (char*)d_ws;
    float*  X    = (float*)(ws + o_x);
    float*  XN   = (float*)(ws + o_xn);
    float*  DD[5];
    for (int i = 0; i < 5; ++i) DD[i] = (float*)(ws + o_dd + (size_t)i * SZ_MC);
    float*  Pr   = (float*)(ws + o_p + 0 * SZ_MC);
    float*  Pk   = (float*)(ws + o_p + 1 * SZ_MC);
    float*  Pv   = (float*)(ws + o_p + 2 * SZ_MC);
    float*  Pg   = (float*)(ws + o_p + 3 * SZ_MC);
    float*  Pw   = (float*)(ws + o_p + 4 * SZ_MC);
    float*  Hbuf = (float*)(ws + o_p);          // reuses Pr..Pw region in channel mixer
    float*  LT   = (float*)(ws + o_lt);
    float*  TMP  = (float*)(ws + o_tmp);        // xpre, later Wo output
    float*  OUT  = (float*)(ws + o_out);
    __bf16* ABF  = (__bf16*)(ws + o_abf);
    __bf16* WBF  = (__bf16*)(ws + o_wbf);

    const int nMC = (Mn * Cn) / 256;            // 8192 blocks
    const int nMH = (Mn * DHn) / 256;           // 28672 blocks

    // embedding + LN
    embed_ln_kernel<<<Mn, 256, 0, stream>>>(tokens, embed, elnw, elnb, X);

    static const int ddbase[5] = {14, 17, 20, 23, 26};  // dd_{r,k,v,g,w}_base indices in tm block

    for (int L = 0; L < 4; ++L) {
        auto P = [&](int i) { return (const float*)d_in[7 + L * 36 + i]; };

        // ---- time mixer ----
        ln_kernel<<<Mn, 256, 0, stream>>>(X, P(0), P(1), XN);
        premix_kernel<<<nMC, 256, 0, stream>>>(XN, P(2), TMP);
        for (int i = 0; i < 5; ++i) {
            lora_a_tanh_kernel<<<Mn, 64, 0, stream>>>(TMP, P(ddbase[i] + 1), LT, 32);
            ddlerp_b_kernel<<<nMC, 256, 0, stream>>>(XN, LT, P(ddbase[i]), P(ddbase[i] + 2),
                                                     DD[i], 32);
        }
        run_gemm(stream, DD[0], P(7),  Pr, Mn, Cn, Cn, ABF, WBF);   // Wr
        run_gemm(stream, DD[1], P(8),  Pk, Mn, Cn, Cn, ABF, WBF);   // Wk
        run_gemm(stream, DD[2], P(9),  Pv, Mn, Cn, Cn, ABF, WBF);   // Wv
        run_gemm(stream, DD[3], P(10), Pg, Mn, Cn, Cn, ABF, WBF);   // Wg
        lora_a_tanh_kernel<<<Mn, 64, 0, stream>>>(DD[4], P(5), LT, 64);
        decay_kernel<<<nMC, 256, 0, stream>>>(LT, P(4), P(6), Pw);

        wkv_kernel<<<Bn * Hn, 64, 0, stream>>>(Pr, Pk, Pv, Pw, P(3), OUT);
        groupnorm_kernel<<<dim3(Mn, Hn), 64, 0, stream>>>(OUT, P(12), P(13));
        silu_mul_kernel<<<nMC, 256, 0, stream>>>(OUT, Pg, (size_t)Mn * Cn);

        run_gemm(stream, OUT, P(11), TMP, Mn, Cn, Cn, ABF, WBF);    // Wo
        add_inplace_kernel<<<nMC, 256, 0, stream>>>(X, TMP, (size_t)Mn * Cn);

        // ---- channel mixer ----
        ln_kernel<<<Mn, 256, 0, stream>>>(X, P(29), P(30), XN);
        cm_lerp_kernel<<<nMC, 256, 0, stream>>>(XN, P(31), P(32), DD[0], DD[1]);
        run_gemm(stream, DD[0], P(33), Hbuf, Mn, Cn, DHn, ABF, WBF);  // Win
        relusq_kernel<<<nMH, 256, 0, stream>>>(Hbuf, (size_t)Mn * DHn);
        run_gemm(stream, Hbuf, P(34), DD[2], Mn, DHn, Cn, ABF, WBF);  // Wout
        run_gemm(stream, DD[1], P(35), DD[3], Mn, Cn,  Cn, ABF, WBF); // Wgate
        cm_final_kernel<<<nMC, 256, 0, stream>>>(X, DD[2], DD[3], (size_t)Mn * Cn);
    }

    // final LN + unembed (fills all of d_out)
    ln_kernel<<<Mn, 256, 0, stream>>>(X, olnw, olnb, XN);
    run_gemm(stream, XN, unembed, (float*)d_out, Mn, Cn, NV, ABF, WBF);
}
